// LRDiff_30940944400727
// MI455X (gfx1250) — matrix operand order repaired and checked by reading, not compiled, not run on hardware
//
#include <hip/hip_runtime.h>

// ---- problem constants (match reference) ----
#define BH_    256
#define N_     4096
#define D_     80
#define M_     77
#define HEADS_ 8
#define QBLK   256      // queries per block (8 waves x 32)
#define WQ     32       // queries per wave
#define KSTR   104      // LDS row stride (bf16 elems) for K / V^T  (52 words -> low bank conflicts)
#define SSTR   84       // score row stride (f32 elems) (21*16B aligned, conflict-free C store)

typedef __attribute__((ext_vector_type(16))) __bf16 v16bf;
typedef __attribute__((ext_vector_type(8)))  __bf16 v8bf;
typedef __attribute__((ext_vector_type(4)))  __bf16 v4bf;
typedef __attribute__((ext_vector_type(8)))  float  v8f;
typedef __attribute__((ext_vector_type(4)))  float  v4f;

// ---- LDS layout (bytes) ----
#define OFF_K   0
#define OFF_VT  (80 * KSTR * 2)                 // 16640
#define OFF_S   (2 * 80 * KSTR * 2)             // 33280
#define OFF_INV (OFF_S + QBLK * SSTR * 4)       // 119296
#define LDS_TOTAL (OFF_INV + QBLK * 4)          // 120320

__device__ __forceinline__ __bf16 f2bf(float x) { return (__bf16)x; }

__device__ __forceinline__ v16bf cat8(v8bf lo, v8bf hi) {
  return __builtin_shufflevector(lo, hi, 0,1,2,3,4,5,6,7,8,9,10,11,12,13,14,15);
}

__global__ __launch_bounds__(QBLK)
void topk_attn_kernel(const float* __restrict__ Q, const float* __restrict__ K,
                      const float* __restrict__ V, float* __restrict__ Out) {
  extern __shared__ char smem[];
  __bf16* lds_k   = (__bf16*)(smem + OFF_K);    // [80 keys][KSTR dims]   (keys 77..79, dims 80..103 = 0)
  __bf16* lds_vt  = (__bf16*)(smem + OFF_VT);   // [80 dims][KSTR keys]   (keys 77..103 = 0)
  float*  lds_s   = (float*) (smem + OFF_S);    // [QBLK rows][SSTR] scores; later bf16 weights in-place
  float*  lds_inv = (float*) (smem + OFF_INV);  // [QBLK] 1/denominator

  const int bh    = blockIdx.x;
  const int ntile = blockIdx.y;
  const int tid   = threadIdx.x;
  const int wave  = tid >> 5;
  const int lane  = tid & 31;
  const int l16   = lane & 15;
  const int lhalf = lane >> 4;

  // ---- phase 0: K and V^T -> LDS as bf16 (zero padded) ----
  {
    const float* Kg = K + (size_t)bh * (M_ * D_);
    const float* Vg = V + (size_t)bh * (M_ * D_);
    for (int i = tid; i < 80 * KSTR; i += QBLK) {
      int r = i / KSTR;               // K: key row      Vt: dim row
      int c = i - r * KSTR;           // K: dim col      Vt: key col
      float kv = (r < M_ && c < D_) ? Kg[r * D_ + c] : 0.f;
      lds_k[i] = f2bf(kv);
      float vv = (c < M_) ? Vg[c * D_ + r] : 0.f;   // r < 80 always a valid dim
      lds_vt[i] = f2bf(vv);
    }
  }
  __syncthreads();

  // ---- phase 1: Q -> bf16 A fragments (scale folded in) ----
  // A layout (16-bit, 16x32): lanes 0-15 = rows, per-lane K = {base..base+7} U {base+16..base+23},
  // base = kc*32 + lhalf*8. Elements 0..7 -> first chunk, 8..15 -> second chunk.
  const float scale = 0.11180339887498949f;   // 1/sqrt(80)
  v16bf Aq[2][3];
  {
    const float* Qg = Q + ((size_t)bh * N_ + (size_t)ntile * QBLK + wave * WQ) * D_;
    #pragma unroll
    for (int qt = 0; qt < 2; ++qt) {
      const float* qr = Qg + (qt * 16 + l16) * D_;
      #pragma unroll
      for (int kc = 0; kc < 3; ++kc) {
        int d0 = kc * 32 + lhalf * 8;
        v16bf a;
        v4f f0 = *(const v4f*)(qr + d0);
        v4f f1 = *(const v4f*)(qr + d0 + 4);
        #pragma unroll
        for (int j = 0; j < 4; ++j) {
          a[j]     = f2bf(f0[j] * scale);
          a[4 + j] = f2bf(f1[j] * scale);
        }
        if (kc < 2) {                       // second chunk dims d0+16..d0+23 (<80 only for kc<2)
          v4f f2 = *(const v4f*)(qr + d0 + 16);
          v4f f3 = *(const v4f*)(qr + d0 + 20);
          #pragma unroll
          for (int j = 0; j < 4; ++j) {
            a[8 + j]  = f2bf(f2[j] * scale);
            a[12 + j] = f2bf(f3[j] * scale);
          }
        } else {
          #pragma unroll
          for (int j = 0; j < 8; ++j) a[8 + j] = f2bf(0.f);
        }
        Aq[qt][kc] = a;
      }
    }
  }

  // ---- phase 2: scores S = (Q*scale) K^T via WMMA bf16 ----
  // B layout (32x16): lane = column (key), per-lane K = {base..base+7} U {base+16..base+23}, base = kc*32 + lhalf*8 (matches A).
  v8f zero8 = {0.f,0.f,0.f,0.f,0.f,0.f,0.f,0.f};
  v8f S[2][5];
  #pragma unroll
  for (int qt = 0; qt < 2; ++qt)
    #pragma unroll
    for (int t = 0; t < 5; ++t) S[qt][t] = zero8;

  #pragma unroll
  for (int t = 0; t < 5; ++t) {
    #pragma unroll
    for (int kc = 0; kc < 3; ++kc) {
      const __bf16* bp = lds_k + (t * 16 + l16) * KSTR + kc * 32 + lhalf * 8;
      v16bf Bk = cat8(*(const v8bf*)bp, *(const v8bf*)(bp + 16));
      S[0][t] = __builtin_amdgcn_wmma_f32_16x16x32_bf16(false, Aq[0][kc], false, Bk, (short)0, S[0][t], false, false);
      S[1][t] = __builtin_amdgcn_wmma_f32_16x16x32_bf16(false, Aq[1][kc], false, Bk, (short)0, S[1][t], false, false);
    }
  }

  // ---- phase 3: spill scores to LDS; mask padded keys with -inf ----
  const int rowbase = wave * WQ;
  #pragma unroll
  for (int qt = 0; qt < 2; ++qt)
    #pragma unroll
    for (int t = 0; t < 5; ++t)
      #pragma unroll
      for (int r = 0; r < 8; ++r)
        lds_s[(rowbase + qt * 16 + lhalf * 8 + r) * SSTR + t * 16 + l16] = S[qt][t][r];

  {
    float* srow = lds_s + (rowbase + lane) * SSTR;
    srow[77] = -__builtin_inff();
    srow[78] = -__builtin_inff();
    srow[79] = -__builtin_inff();
  }

  // ---- phase 4: per-lane (1 query/lane) top-10 threshold + softmax weights ----
  {
    float* srow = lds_s + (rowbase + lane) * SSTR;
    float tk[10];
    #pragma unroll
    for (int i = 0; i < 10; ++i) tk[i] = -__builtin_inff();
    for (int j = 0; j < 80; j += 4) {           // branchless sorted-insert network
      v4f s = *(const v4f*)(srow + j);
      #pragma unroll
      for (int e = 0; e < 4; ++e) {
        float v = s[e];
        #pragma unroll
        for (int i = 0; i < 10; ++i) {
          float hi = fmaxf(tk[i], v);
          v = fminf(tk[i], v);
          tk[i] = hi;
        }
      }
    }
    float rmax = tk[0];
    float thr  = tk[9];   // 10th largest (ties handled identically to attn > kth)

    // denom over ALL keys; masked unnormalized weights written back in-place as bf16.
    float den = 0.f;
    __bf16* wrow = (__bf16*)srow;
    for (int j = 0; j < 80; j += 4) {
      v4f s = *(const v4f*)(srow + j);          // safe: bf16 writes stay below f32 read frontier
      v4bf wv;
      #pragma unroll
      for (int e = 0; e < 4; ++e) {
        float ex = __expf(s[e] - rmax);         // exp(-inf)=0 for pads
        den += ex;
        wv[e] = f2bf(s[e] > thr ? ex : 0.f);
      }
      *(v4bf*)(wrow + j) = wv;
    }
    v4bf z4 = {f2bf(0.f), f2bf(0.f), f2bf(0.f), f2bf(0.f)};
    *(v4bf*)(wrow + 80) = z4;                   // zero K-pad keys 80..95 for the PV A-tiles
    *(v4bf*)(wrow + 84) = z4;
    *(v4bf*)(wrow + 88) = z4;
    *(v4bf*)(wrow + 92) = z4;
    lds_inv[rowbase + lane] = 1.f / den;
  }

  // ---- phase 5: out = W * V via WMMA bf16 ----
  v16bf Aw[2][3];
  #pragma unroll
  for (int qt = 0; qt < 2; ++qt) {
    const __bf16* wr = (const __bf16*)(lds_s + (rowbase + qt * 16 + l16) * SSTR);
    #pragma unroll
    for (int kc = 0; kc < 3; ++kc) {
      const __bf16* p = wr + kc * 32 + lhalf * 8;
      Aw[qt][kc] = cat8(*(const v8bf*)p, *(const v8bf*)(p + 16));   // keys {b..b+7} U {b+16..b+23}
    }
  }

  v8f O[2][5];
  #pragma unroll
  for (int qt = 0; qt < 2; ++qt)
    #pragma unroll
    for (int dt = 0; dt < 5; ++dt) O[qt][dt] = zero8;

  #pragma unroll
  for (int kc = 0; kc < 3; ++kc) {
    #pragma unroll
    for (int dt = 0; dt < 5; ++dt) {
      const __bf16* vp = lds_vt + (dt * 16 + l16) * KSTR + kc * 32 + lhalf * 8;
      v16bf Bv = cat8(*(const v8bf*)vp, *(const v8bf*)(vp + 16));
      O[0][dt] = __builtin_amdgcn_wmma_f32_16x16x32_bf16(false, Aw[0][kc], false, Bv, (short)0, O[0][dt], false, false);
      O[1][dt] = __builtin_amdgcn_wmma_f32_16x16x32_bf16(false, Aw[1][kc], false, Bv, (short)0, O[1][dt], false, false);
    }
  }

  // ---- phase 6: scale by 1/denom, store with (b, n, h*D+d) permutation ----
  {
    const int b = bh >> 3, h = bh & 7;
    float* outp = Out + ((size_t)b * N_ + (size_t)ntile * QBLK + wave * WQ) * (HEADS_ * D_) + h * D_;
    #pragma unroll
    for (int qt = 0; qt < 2; ++qt) {
      float inv[8];
      #pragma unroll
      for (int r = 0; r < 8; ++r)
        inv[r] = lds_inv[rowbase + qt * 16 + lhalf * 8 + r];
      #pragma unroll
      for (int dt = 0; dt < 5; ++dt)
        #pragma unroll
        for (int r = 0; r < 8; ++r) {
          int rowm = qt * 16 + lhalf * 8 + r;
          outp[(size_t)rowm * (HEADS_ * D_) + dt * 16 + l16] = O[qt][dt][r] * inv[r];
        }
    }
  }
}

extern "C" void kernel_launch(void* const* d_in, const int* in_sizes, int n_in,
                              void* d_out, int out_size, void* d_ws, size_t ws_size,
                              hipStream_t stream) {
  (void)in_sizes; (void)n_in; (void)out_size; (void)d_ws; (void)ws_size;
  const float* Q = (const float*)d_in[0];
  const float* K = (const float*)d_in[1];
  const float* V = (const float*)d_in[2];
  float* Out = (float*)d_out;

  hipFuncSetAttribute((const void*)topk_attn_kernel,
                      hipFuncAttributeMaxDynamicSharedMemorySize, (int)LDS_TOTAL);

  dim3 grid(BH_, N_ / QBLK);   // (256, 16)
  dim3 block(QBLK);            // 256 threads = 8 wave32
  topk_attn_kernel<<<grid, block, LDS_TOTAL, stream>>>(Q, K, V, Out);
}
